// coESN_RFF_75299366633679
// MI455X (gfx1250) — compile-verified, matrix operand order checked by reading, not run on hardware
//
#include <hip/hip_runtime.h>
#include <math.h>

// ---------------------------------------------------------------------------
// coESN + RFF recurrence for MI455X (gfx1250, wave32, WMMA bf16 16x16x32)
//
//   B=32, L=2048, I=64, H=512, HALF=256
//   8 persistent workgroups, each owns a 64-column slice of H (and of z).
//   Weights live in LDS as pre-swizzled WMMA B-fragments (bf16).
//   hy state exchanged each step through a double-buffered bf16 buffer in
//   global memory (32 KB -> L2/WGP$ resident), one grid barrier per step.
// ---------------------------------------------------------------------------

typedef __bf16 bf16_t;
typedef bf16_t v16bf __attribute__((ext_vector_type(16)));
typedef float  v8f   __attribute__((ext_vector_type(8)));

namespace {
constexpr int   kB   = 32;
constexpr int   kL   = 2048;
constexpr int   kI   = 64;
constexpr int   kH   = 512;
constexpr float kDT  = 0.05f;
constexpr int   kNWG    = 8;            // workgroups (H-slices)
constexpr int   kSlice  = kH / kNWG;    // 64 columns per WG
constexpr int   kNT     = kSlice / 16;  // 4 N-tiles per WG
constexpr int   kKTH    = kH / 32;      // 16 K-tiles over H
constexpr int   kKTI    = kI / 32;      // 2  K-tiles over I
constexpr int   kThreads = 256;         // 8 wave32 = (4 nt) x (2 mt)
// mu = (zsum * 1/sqrt(HALF)) / L ;  1/sqrt(256) = 0.0625
constexpr float kOutScale = 0.0625f / 2048.0f;
}

union FragU { v16bf v; uint4 u[2]; };

// --------------------------- grid barrier ----------------------------------
__device__ __forceinline__ void grid_barrier(unsigned* cnt, unsigned* gen,
                                             unsigned nblk) {
  __syncthreads();
  if (threadIdx.x == 0) {
    __threadfence();  // release: make hy stores visible device-wide
    unsigned g = __atomic_load_n(gen, __ATOMIC_RELAXED);
    if (__atomic_fetch_add(cnt, 1u, __ATOMIC_ACQ_REL) == nblk - 1u) {
      __atomic_store_n(cnt, 0u, __ATOMIC_RELAXED);
      __atomic_fetch_add(gen, 1u, __ATOMIC_RELEASE);
    } else {
      while (__atomic_load_n(gen, __ATOMIC_ACQUIRE) == g) {
        __builtin_amdgcn_s_sleep(1);
      }
    }
    __threadfence();  // acquire: invalidate stale lines
  }
  __syncthreads();
}

// ----------------- A-fragment gather from row-major bf16 -------------------
// A 16x32 bf16 layout: lane l -> row (l&15); kbase = (l>>4)*8;
// elems 0..7 -> k = kt*32 + kbase + j ; elems 8..15 -> k = kt*32+16+kbase+(j-8)
__device__ __forceinline__ v16bf load_a_hy(const bf16_t* hy, int mt, int kt,
                                           int lane) {
  const int m  = mt * 16 + (lane & 15);
  const int kb = kt * 32 + ((lane >> 4) * 8);
  FragU f;
  f.u[0] = *reinterpret_cast<const uint4*>(hy + m * kH + kb);
  f.u[1] = *reinterpret_cast<const uint4*>(hy + m * kH + kb + 16);
  return f.v;
}

__device__ __forceinline__ v16bf load_a_x(const float* x, int t, int mt,
                                          int kt, int lane) {
  const int b  = mt * 16 + (lane & 15);
  const int kb = kt * 32 + ((lane >> 4) * 8);
  const float* p = x + ((size_t)b * kL + t) * kI;
  const float4 f0 = *reinterpret_cast<const float4*>(p + kb);
  const float4 f1 = *reinterpret_cast<const float4*>(p + kb + 4);
  const float4 g0 = *reinterpret_cast<const float4*>(p + kb + 16);
  const float4 g1 = *reinterpret_cast<const float4*>(p + kb + 20);
  v16bf r;
  r[0]  = (bf16_t)f0.x; r[1]  = (bf16_t)f0.y; r[2]  = (bf16_t)f0.z; r[3]  = (bf16_t)f0.w;
  r[4]  = (bf16_t)f1.x; r[5]  = (bf16_t)f1.y; r[6]  = (bf16_t)f1.z; r[7]  = (bf16_t)f1.w;
  r[8]  = (bf16_t)g0.x; r[9]  = (bf16_t)g0.y; r[10] = (bf16_t)g0.z; r[11] = (bf16_t)g0.w;
  r[12] = (bf16_t)g1.x; r[13] = (bf16_t)g1.y; r[14] = (bf16_t)g1.z; r[15] = (bf16_t)g1.w;
  return r;
}

__device__ __forceinline__ v16bf load_b_lds(const uint4* lds, int frag,
                                            int lane) {
  FragU f;
  f.u[0] = lds[(frag * 32 + lane) * 2 + 0];
  f.u[1] = lds[(frag * 32 + lane) * 2 + 1];
  return f.v;
}

// ------------------------------- init --------------------------------------
__global__ void coesn_init_ws(unsigned* ws) {
  // zero barrier words (first 256B) + both hy exchange buffers (2 * 32KB)
  const int nwords = (256 + 2 * kB * kH * (int)sizeof(bf16_t)) / 4;
  for (int i = blockIdx.x * blockDim.x + threadIdx.x; i < nwords;
       i += gridDim.x * blockDim.x)
    ws[i] = 0u;
}

// ------------------------------- main --------------------------------------
__global__ __launch_bounds__(kThreads, 1)
void coesn_rff_kernel(const float* __restrict__ x,
                      const float* __restrict__ x2h,
                      const float* __restrict__ h2h,
                      const float* __restrict__ bias,
                      const float* __restrict__ gamma,
                      const float* __restrict__ eps,
                      const float* __restrict__ wrff,
                      float* __restrict__ out,
                      unsigned char* __restrict__ ws) {
  // LDS: pre-swizzled bf16 B-fragments (frag = kt*kNT + nt, 32 lanes, 2 uint4)
  __shared__ uint4 sh_h2h[kKTH * kNT * 32 * 2];  // 64 KB
  __shared__ uint4 sh_wrf[kKTH * kNT * 32 * 2];  // 64 KB
  __shared__ uint4 sh_x2h[kKTI * kNT * 32 * 2];  //  8 KB

  const int tid  = threadIdx.x;
  const int lane = tid & 31;
  const int wv   = tid >> 5;       // 0..7
  const int nt   = wv & 3;         // N-tile within slice
  const int mt   = wv >> 2;        // M-tile (batch rows 0..15 / 16..31)
  const int wg   = blockIdx.x;     // 0..7
  const int n0   = wg * kSlice;    // hy/z column base of this WG
  const bool useCos = (wg < 4);    // z cols [0,256)=cos, [256,512)=sin
  const int jsl0 = (wg & 3) * kSlice;  // wx column base for W_rff slice

  // ---- stage weights into LDS as B-fragments (one-time) ----
  for (int s = tid; s < kKTH * kNT * 32; s += blockDim.x) {
    const int l    = s & 31;
    const int frag = s >> 5;
    const int fnt  = frag & (kNT - 1);
    const int fkt  = frag / kNT;
    const int kb   = fkt * 32 + ((l >> 4) * 8);
    // h2h slice: B[k][n] = h2h[k*H + n0 + n]
    {
      const int n = n0 + fnt * 16 + (l & 15);
      FragU f;
#pragma unroll
      for (int j = 0; j < 8; ++j) f.v[j]     = (bf16_t)h2h[(kb + j) * kH + n];
#pragma unroll
      for (int j = 0; j < 8; ++j) f.v[8 + j] = (bf16_t)h2h[(kb + 16 + j) * kH + n];
      sh_h2h[s * 2] = f.u[0]; sh_h2h[s * 2 + 1] = f.u[1];
    }
    // W_rff slice: B[k][j] = W_rff[(jsl0+j)*H + k]   (wx = hy @ W_rff^T)
    {
      const int j = jsl0 + fnt * 16 + (l & 15);
      FragU f;
#pragma unroll
      for (int q = 0; q < 8; ++q) f.v[q]     = (bf16_t)wrff[j * kH + kb + q];
#pragma unroll
      for (int q = 0; q < 8; ++q) f.v[8 + q] = (bf16_t)wrff[j * kH + kb + 16 + q];
      sh_wrf[s * 2] = f.u[0]; sh_wrf[s * 2 + 1] = f.u[1];
    }
  }
  for (int s = tid; s < kKTI * kNT * 32; s += blockDim.x) {
    const int l    = s & 31;
    const int frag = s >> 5;
    const int fnt  = frag & (kNT - 1);
    const int fkt  = frag / kNT;
    const int kb   = fkt * 32 + ((l >> 4) * 8);
    const int n    = n0 + fnt * 16 + (l & 15);
    FragU f;
#pragma unroll
    for (int j = 0; j < 8; ++j) f.v[j]     = (bf16_t)x2h[(kb + j) * kH + n];
#pragma unroll
    for (int j = 0; j < 8; ++j) f.v[8 + j] = (bf16_t)x2h[(kb + 16 + j) * kH + n];
    sh_x2h[s * 2] = f.u[0]; sh_x2h[s * 2 + 1] = f.u[1];
  }
  __syncthreads();

  // ---- per-lane column constants (D-layout: column fixed per lane) ----
  const int col = n0 + nt * 16 + (lane & 15);
  const float bias_v  = bias[col];
  const float gamma_v = gamma[col];
  const float eps_v   = eps[col];

  // ---- workspace layout ----
  unsigned* cnt = reinterpret_cast<unsigned*>(ws);          // offset 0
  unsigned* gen = reinterpret_cast<unsigned*>(ws + 128);    // offset 128
  bf16_t* buf0  = reinterpret_cast<bf16_t*>(ws + 256);
  bf16_t* buf1  = buf0 + kB * kH;

  // ---- f32 register state (one 16x16 D-tile per wave) ----
  v8f hz  = {};
  v8f hyf = {};
  v8f zs  = {};

  for (int t = 0; t < kL; ++t) {
    const bf16_t* cur  = (t & 1) ? buf1 : buf0;   // hy_t  (full, bf16)
    bf16_t*       next = (t & 1) ? buf0 : buf1;   // hy_{t+1}

    if (t + 1 < kL)
      __builtin_prefetch(x + ((size_t)(mt * 16 + (lane & 15)) * kL + t + 1) * kI, 0, 0);

    // -- pre-activation: C = hy_t @ h2h_slice + x_t @ x2h_slice --
    v8f c = {};
#pragma unroll
    for (int kt = 0; kt < kKTH; ++kt) {
      const v16bf b = load_b_lds(sh_h2h, kt * kNT + nt, lane);
      const v16bf a = load_a_hy(cur, mt, kt, lane);
      c = __builtin_amdgcn_wmma_f32_16x16x32_bf16(false, a, false, b,
                                                  (short)0, c, false, false);
    }
#pragma unroll
    for (int kt = 0; kt < kKTI; ++kt) {
      const v16bf b = load_b_lds(sh_x2h, kt * kNT + nt, lane);
      const v16bf a = load_a_x(x, t, mt, kt, lane);
      c = __builtin_amdgcn_wmma_f32_16x16x32_bf16(false, a, false, b,
                                                  (short)0, c, false, false);
    }

    // -- elementwise ODE update (f32) + publish hy_{t+1} slice (bf16) --
#pragma unroll
    for (int r = 0; r < 8; ++r) {
      const float pre = tanhf(c[r] + bias_v);
      hz[r]  += kDT * (pre - gamma_v * hyf[r] - eps_v * hz[r]);
      hyf[r] += kDT * hz[r];
      const int m = mt * 16 + r + ((lane >> 4) * 8);
      next[m * kH + col] = (bf16_t)hyf[r];
    }

    grid_barrier(cnt, gen, kNWG);

    // -- RFF projection: wx = hy_{t+1} @ W_rff_slice^T --
    v8f w = {};
#pragma unroll
    for (int kt = 0; kt < kKTH; ++kt) {
      const v16bf b = load_b_lds(sh_wrf, kt * kNT + nt, lane);
      const v16bf a = load_a_hy(next, mt, kt, lane);
      w = __builtin_amdgcn_wmma_f32_16x16x32_bf16(false, a, false, b,
                                                  (short)0, w, false, false);
    }
#pragma unroll
    for (int r = 0; r < 8; ++r)
      zs[r] += useCos ? cosf(w[r]) : sinf(w[r]);
  }

  // ---- epilogue: d_out = [ mu (32x512) | hy (32x512) ] ----
#pragma unroll
  for (int r = 0; r < 8; ++r) {
    const int m = mt * 16 + r + ((lane >> 4) * 8);
    out[m * kH + col]            = zs[r] * kOutScale;
    out[kB * kH + m * kH + col]  = hyf[r];
  }
}

// ---------------------------------------------------------------------------
extern "C" void kernel_launch(void* const* d_in, const int* in_sizes, int n_in,
                              void* d_out, int out_size, void* d_ws,
                              size_t ws_size, hipStream_t stream) {
  (void)in_sizes; (void)n_in; (void)out_size; (void)ws_size;
  const float* x     = (const float*)d_in[0];
  const float* x2h   = (const float*)d_in[1];
  const float* h2h   = (const float*)d_in[2];
  const float* bias  = (const float*)d_in[3];
  const float* gamma = (const float*)d_in[4];
  const float* eps   = (const float*)d_in[5];
  const float* wrff  = (const float*)d_in[6];

  coesn_init_ws<<<64, 256, 0, stream>>>((unsigned*)d_ws);
  coesn_rff_kernel<<<kNWG, kThreads, 0, stream>>>(
      x, x2h, h2h, bias, gamma, eps, wrff, (float*)d_out,
      (unsigned char*)d_ws);
}
